// CodebookConv1D_72181220377030
// MI455X (gfx1250) — matrix operand order repaired and checked by reading, not compiled, not run on hardware
//
#include <hip/hip_runtime.h>

// CodebookConv1D on MI455X (gfx1250):
//   out(8192x8192) = X(8192x2048) @ dequant(W)(2048x8192) + bias
// 275 GFLOP over ~0.34 GB of HBM traffic -> matrix-core bound.
// v_wmma_f32_16x16x32_bf16 (bf16 in, f32 accumulate); codebook-compressed W
// is dequantized straight into LDS (transposed) each K-step. Software
// pipeline: next tile's global loads (X + indices + codeword gathers) are
// issued right after the barrier and overlap the 8 WMMAs; LDS is
// double-buffered so there is a single barrier per K-step.

typedef __attribute__((ext_vector_type(16))) __bf16 v16bf;
typedef __attribute__((ext_vector_type(8)))  float  v8f;

#define IN_F   2048
#define OUT_F  8192
#define M_TOT  8192          // 4 * 2048 rows of x
#define BM     128
#define BN     128
#define BK     32
#define NKT    (IN_F / BK)   // 64 K-steps
#define LSTR   40            // padded LDS row stride (elements): 80B rows keep
                             // 16B alignment and dodge 64-bank conflicts
#define THREADS 256

union FragU  { uint4 u[2]; v16bf v; };      // 32B = one 16x32 bf16 fragment / lane
union Pack16 { __bf16 h[16]; uint4 u[2]; };

__global__ __launch_bounds__(THREADS)
void cbconv_wmma_kernel(const float* __restrict__ X,
                        const float* __restrict__ codebook,
                        const int*   __restrict__ indices,
                        const float* __restrict__ bias,
                        float* __restrict__ Out)
{
    __shared__ __bf16 As[2][BM * LSTR];   // X tile, [m][k], bf16 (double-buffered)
    __shared__ __bf16 Ws[2][BN * LSTR];   // W tile transposed, [n][k], bf16

    const int t    = threadIdx.x;
    const int lane = t & 31;
    const int wv   = t >> 5;           // wave 0..7
    const int wm   = wv >> 1;          // 0..3 : 32-row slab
    const int wn   = wv & 1;           // 0..1 : 64-col slab
    const int lrow = lane & 15;
    const int hi   = lane >> 4;

    const int m0 = blockIdx.y * BM;
    const int n0 = blockIdx.x * BN;

    v8f acc[2][4] = {};

    // staging coordinates (per thread)
    const int arow = t >> 1;           // 0..127 : X tile row
    const int acol = (t & 1) << 4;     // 0 or 16 : X tile col base
    const int nb_l = t & 15;           // block-column within tile (8 n's each)
    const int kl0  = t >> 4;           // 0..15  : k row within tile

    const float* xptr = X + (size_t)(m0 + arow) * IN_F + acol;   // + kt
    const int*   iptr = indices + (n0 >> 3) + nb_l;              // + gk*(OUT_F/8)

    // ---- pipeline registers ----
    float4 xf[4];       // 16 x-values (f32)
    float4 wc[4];       // 2 codewords (8 f32 each)

    // prologue: fetch tile kt = 0
    {
        const float4* s = (const float4*)xptr;
        xf[0] = s[0]; xf[1] = s[1]; xf[2] = s[2]; xf[3] = s[3];
        const int i0 = iptr[(kl0     ) * (OUT_F / 8)];
        const int i1 = iptr[(kl0 + 16) * (OUT_F / 8)];
        const float4* c0 = (const float4*)(codebook + ((size_t)i0 << 3));
        const float4* c1 = (const float4*)(codebook + ((size_t)i1 << 3));
        wc[0] = c0[0]; wc[1] = c0[1]; wc[2] = c1[0]; wc[3] = c1[1];
    }

    for (int it = 0; it < NKT; ++it) {
        const int buf = it & 1;

        // ---- stage pipelined registers -> LDS[buf] (f32 -> bf16) ----
        {
            Pack16 p;
            p.h[0] =(__bf16)xf[0].x; p.h[1] =(__bf16)xf[0].y;
            p.h[2] =(__bf16)xf[0].z; p.h[3] =(__bf16)xf[0].w;
            p.h[4] =(__bf16)xf[1].x; p.h[5] =(__bf16)xf[1].y;
            p.h[6] =(__bf16)xf[1].z; p.h[7] =(__bf16)xf[1].w;
            p.h[8] =(__bf16)xf[2].x; p.h[9] =(__bf16)xf[2].y;
            p.h[10]=(__bf16)xf[2].z; p.h[11]=(__bf16)xf[2].w;
            p.h[12]=(__bf16)xf[3].x; p.h[13]=(__bf16)xf[3].y;
            p.h[14]=(__bf16)xf[3].z; p.h[15]=(__bf16)xf[3].w;
            uint4* dst = (uint4*)&As[buf][arow * LSTR + acol];
            dst[0] = p.u[0];
            dst[1] = p.u[1];
        }
        #pragma unroll
        for (int rep = 0; rep < 2; ++rep) {
            // transpose-scatter one 8-wide codeword block: Ws[n][k]
            const int k_l = kl0 + rep * 16;
            const float4 a0 = wc[rep * 2];
            const float4 a1 = wc[rep * 2 + 1];
            __bf16* w = &Ws[buf][(nb_l * 8) * LSTR + k_l];
            w[0*LSTR] = (__bf16)a0.x;  w[1*LSTR] = (__bf16)a0.y;
            w[2*LSTR] = (__bf16)a0.z;  w[3*LSTR] = (__bf16)a0.w;
            w[4*LSTR] = (__bf16)a1.x;  w[5*LSTR] = (__bf16)a1.y;
            w[6*LSTR] = (__bf16)a1.z;  w[7*LSTR] = (__bf16)a1.w;
        }

        __syncthreads();   // single barrier per K-step (double-buffered LDS)

        // ---- issue next tile's global loads; they overlap the WMMAs ----
        if (it + 1 < NKT) {
            const int kt2 = (it + 1) * BK;
            const float4* s = (const float4*)(xptr + kt2);
            xf[0] = s[0]; xf[1] = s[1]; xf[2] = s[2]; xf[3] = s[3];
            const int i0 = iptr[(kt2 + kl0     ) * (OUT_F / 8)];
            const int i1 = iptr[(kt2 + kl0 + 16) * (OUT_F / 8)];
            const float4* c0 = (const float4*)(codebook + ((size_t)i0 << 3));
            const float4* c1 = (const float4*)(codebook + ((size_t)i1 << 3));
            wc[0] = c0[0]; wc[1] = c0[1]; wc[2] = c1[0]; wc[3] = c1[1];
        }

        // ---- load fragments (ISA 16-bit A / B VGPR layouts) ----
        FragU a[2], b[4];
        #pragma unroll
        for (int i = 0; i < 2; ++i) {
            // A 16x32: lane<16 holds K 0-7 & 16-23, lane>=16 holds K 8-15 & 24-31
            const __bf16* p = &As[buf][(wm*32 + i*16 + lrow) * LSTR + hi*8];
            a[i].u[0] = *(const uint4*)p;
            a[i].u[1] = *(const uint4*)(p + 16);
        }
        #pragma unroll
        for (int j = 0; j < 4; ++j) {
            // B 32x16 (from Ws = W^T): lane<16 holds K 0-7 & 16-23, lane>=16 holds K 8-15 & 24-31
            const __bf16* p = &Ws[buf][(wn*64 + j*16 + lrow) * LSTR + hi*8];
            b[j].u[0] = *(const uint4*)p;
            b[j].u[1] = *(const uint4*)(p + 16);
        }

        // ---- 8 WMMAs per wave per K-step ----
        #pragma unroll
        for (int i = 0; i < 2; ++i)
            #pragma unroll
            for (int j = 0; j < 4; ++j)
                acc[i][j] = __builtin_amdgcn_wmma_f32_16x16x32_bf16(
                    false, a[i].v, false, b[j].v,
                    (short)0, acc[i][j], false, false);
    }

    // ---- epilogue: bias + store (C layout: lane = N, VGPR r = M row) ----
    #pragma unroll
    for (int j = 0; j < 4; ++j) {
        const int ng = n0 + wn*64 + j*16 + lrow;
        const float bv = bias[ng];
        #pragma unroll
        for (int i = 0; i < 2; ++i) {
            const int mg = m0 + wm*32 + i*16 + hi*8;
            #pragma unroll
            for (int r = 0; r < 8; ++r)
                Out[(size_t)(mg + r) * OUT_F + ng] = acc[i][j][r] + bv;
        }
    }
}

extern "C" void kernel_launch(void* const* d_in, const int* in_sizes, int n_in,
                              void* d_out, int out_size, void* d_ws, size_t ws_size,
                              hipStream_t stream) {
    const float* X        = (const float*)d_in[0];
    const float* codebook = (const float*)d_in[1];
    const int*   indices  = (const int*)  d_in[2];
    const float* bias     = (const float*)d_in[3];
    float*       out      = (float*)d_out;
    (void)in_sizes; (void)n_in; (void)out_size; (void)d_ws; (void)ws_size;

    dim3 grid(OUT_F / BN, M_TOT / BM);   // 64 x 64 tiles
    cbconv_wmma_kernel<<<grid, THREADS, 0, stream>>>(X, codebook, indices, bias, out);
}
